// SchNet_7928509628805
// MI455X (gfx1250) — matrix operand order repaired and checked by reading, not compiled, not run on hardware
//
#include <hip/hip_runtime.h>
#include <hip/hip_bf16.h>
#include <math.h>

typedef __attribute__((ext_vector_type(16))) _Float16 v16h;
typedef __attribute__((ext_vector_type(8)))  _Float16 v8h;
typedef __attribute__((ext_vector_type(8)))  float    v8f;

#define NN 10000
#define EE 320000
#define GG 100
#define DIM 128
#define NF 128
#define NG 50
#define NLAYERS 6

// Fragment = 512 halves (1 KB): [chunk(2)][lane(32)][8 halves]
#define FRAG_H 512

__device__ __forceinline__ float sspf(float x) {
    float sp = (x > 20.0f) ? x : log1pf(__expf(x));
    return sp - 0.69314718056f;
}

// ---- swizzle index math (matches ISA 7.12.2 VGPR layouts) ----
// A fragment 16x32 (MxK): lanes 0-15 hold M=lane, K=0..7 & 16..23;
// lanes 16-31 hold M=lane-16, K=8..15 & 24..31.
__device__ __forceinline__ int a_swz_idx(int m, int k) {
    int lane = m + ((k & 8) ? 16 : 0);
    int chunk = (k & 16) ? 1 : 0;
    return chunk * 256 + lane * 8 + (k & 7);
}
// B fragment 32x16 (KxN): lanes 0-15: N=lane, K=0..7 & 16..23; lanes 16-31: N=lane-16, K=8..15 & 24..31.
__device__ __forceinline__ int b_swz_idx(int k, int n) {
    int lane = n + ((k & 8) ? 16 : 0);
    int chunk = (k & 16) ? 1 : 0;
    return chunk * 256 + lane * 8 + (k & 7);
}

// Load one fragment (A from LDS or B from global): two 16B vector loads per lane.
__device__ __forceinline__ v16h load_frag(const _Float16* __restrict__ base) {
    int lane = threadIdx.x & 31;
    v8h lo = *(const v8h*)(base + lane * 8);
    v8h hi = *(const v8h*)(base + 256 + lane * 8);
    return __builtin_shufflevector(lo, hi, 0, 1, 2, 3, 4, 5, 6, 7,
                                   8, 9, 10, 11, 12, 13, 14, 15);
}

__device__ __forceinline__ v8f wmma_f16(v16h a, v16h b, v8f c) {
    return __builtin_amdgcn_wmma_f32_16x16x32_f16(
        false, a, false, b, (short)0, c, false, false);
}

// Stage a 64x128 f32 tile into swizzled f16 LDS fragments; 8 halves per b128 store.
__device__ __forceinline__ void stage_a_tile(const float* __restrict__ in, int row0,
                                             int nrows, _Float16* __restrict__ sA) {
    int tid = threadIdx.x;
    for (int g = tid; g < 64 * 16; g += 128) {
        int r = g >> 4;
        int c8 = (g & 15) * 8;          // 8-aligned column group
        int gr = row0 + r;
        const float* src = in + (size_t)gr * DIM + c8;
        v8h hv;
#pragma unroll
        for (int u = 0; u < 8; ++u)
            hv[u] = (_Float16)((gr < nrows) ? src[u] : 0.0f);
        *(v8h*)&sA[((r >> 4) * 4 + (c8 >> 5)) * FRAG_H + a_swz_idx(r & 15, c8 & 31)] = hv;
    }
}

// ---------------- weight prep: f32 [K][128] -> swizzled f16 fragments ----------------
__global__ void swz_weight_kernel(const float* __restrict__ in,
                                  _Float16* __restrict__ out, int K, int nmats) {
    int i = blockIdx.x * blockDim.x + threadIdx.x;
    int per = K * 128;
    int total = nmats * per;
    if (i >= total) return;
    int mat = i / per;
    int k = (i / 128) % K;
    int n = i % 128;
    int ks = k >> 5, kk = k & 31, nt = n >> 4, nn = n & 15;
    int fragsPerMat = (K >> 5) * 8;
    out[(size_t)mat * fragsPerMat * FRAG_H + (size_t)(ks * 8 + nt) * FRAG_H +
        b_swz_idx(kk, nn)] = (_Float16)in[i];
}

// fw1: [6][50][128] -> swizzled [6][2*8 frags], K padded 50->64 with zeros
__global__ void swz_fw1_kernel(const float* __restrict__ in,
                               _Float16* __restrict__ out) {
    int i = blockIdx.x * blockDim.x + threadIdx.x;
    int total = NLAYERS * 64 * 128;
    if (i >= total) return;
    int l = i / (64 * 128);
    int k = (i / 128) % 64;
    int n = i % 128;
    float v = (k < NG) ? in[(l * NG + k) * 128 + n] : 0.0f;
    int ks = k >> 5, kk = k & 31, nt = n >> 4, nn = n & 15;
    out[(size_t)l * 16 * FRAG_H + (size_t)(ks * 8 + nt) * FRAG_H +
        b_swz_idx(kk, nn)] = (_Float16)v;
}

// ---------------- node GEMM: out[N,128] = in[N,128] @ W(swz 128x128) ----------------
__global__ void __launch_bounds__(128)
node_gemm(const float* __restrict__ in, const _Float16* __restrict__ W,
          float* __restrict__ out, int nrows) {
    __shared__ __attribute__((aligned(16))) _Float16 sA[16 * FRAG_H]; // 4 waves x 4 ks
    int tid = threadIdx.x;
    int row0 = blockIdx.x * 64;
    stage_a_tile(in, row0, nrows, sA);
    __syncthreads();
    int wave = tid >> 5, lane = tid & 31;
    int col_lo = lane & 15;
    int rsub = (lane & 16) ? 8 : 0;
    const _Float16* aw = sA + wave * 4 * FRAG_H;
    v8f acc[8] = {};
#pragma unroll
    for (int ks = 0; ks < 4; ++ks) {
        v16h a = load_frag(aw + ks * FRAG_H);
#pragma unroll
        for (int nt = 0; nt < 8; ++nt) {
            v16h b = load_frag(W + (ks * 8 + nt) * FRAG_H);
            acc[nt] = wmma_f16(a, b, acc[nt]);
        }
    }
#pragma unroll
    for (int nt = 0; nt < 8; ++nt) {
        int col = nt * 16 + col_lo;
#pragma unroll
        for (int i = 0; i < 8; ++i) {
            int gr = row0 + wave * 16 + rsub + i;
            if (gr < nrows) out[gr * NF + col] = acc[nt][i];
        }
    }
}

// ------- fused edge kernel: basis -> GEMM1 -> ssp -> GEMM2 -> *C -> gather/scatter -------
__global__ void __launch_bounds__(128)
edge_kernel(const float* __restrict__ dist, const int* __restrict__ didx,
            const _Float16* __restrict__ fw1h, const float* __restrict__ fb1,
            const _Float16* __restrict__ fw2h, const float* __restrict__ fb2,
            const float* __restrict__ hf, float* __restrict__ agg) {
    __shared__ __attribute__((aligned(16))) _Float16 sBasis[8 * FRAG_H];  // 4 waves x 2 ks
    __shared__ __attribute__((aligned(16))) _Float16 sT1[16 * FRAG_H];    // 4 waves x 4 ks
    __shared__ float sW[64 * 128];
    __shared__ float sC[64];
    __shared__ int   sI[64], sJ[64];
    int tid = threadIdx.x;
    int e0 = blockIdx.x * 64;

    if (tid < 64) {
        int ge = e0 + tid;
        float d = dist[ge];
        float c = 0.5f * (__cosf(d * 0.31415926535f) + 1.0f);
        sC[tid] = (d < 10.0f) ? c : 0.0f;
        sI[tid] = didx[ge];        // ind_i (scatter)
        sJ[tid] = didx[EE + ge];   // ind_j (gather)
    }
    // Gaussian basis -> swizzled A fragments; 8 gaussians per b128 store
    const float step = 10.0f / 49.0f;
    const float coeff = -0.5f / (step * step);
    for (int g = tid; g < 64 * 8; g += 128) {
        int e = g >> 3;
        int k8 = (g & 7) * 8;
        float d = dist[e0 + e];
        v8h hv;
#pragma unroll
        for (int u = 0; u < 8; ++u) {
            int k = k8 + u;
            float v = 0.0f;
            if (k < NG) {
                float t = d - step * (float)k;
                v = __expf(coeff * t * t);
            }
            hv[u] = (_Float16)v;
        }
        *(v8h*)&sBasis[((e >> 4) * 2 + (k8 >> 5)) * FRAG_H + a_swz_idx(e & 15, k8 & 31)] = hv;
    }
    __syncthreads();

    int wave = tid >> 5, lane = tid & 31;
    int col_lo = lane & 15;
    int rsub = (lane & 16) ? 8 : 0;

    // GEMM1: [16,64] x fw1[64,128]
    v8f acc[8] = {};
#pragma unroll
    for (int ks = 0; ks < 2; ++ks) {
        v16h a = load_frag(sBasis + (wave * 2 + ks) * FRAG_H);
#pragma unroll
        for (int nt = 0; nt < 8; ++nt) {
            v16h b = load_frag(fw1h + (ks * 8 + nt) * FRAG_H);
            acc[nt] = wmma_f16(a, b, acc[nt]);
        }
    }
    // ssp -> swizzled A layout for GEMM2
#pragma unroll
    for (int nt = 0; nt < 8; ++nt) {
        int col = nt * 16 + col_lo;          // K index of GEMM2
        float bi = fb1[col];
#pragma unroll
        for (int i = 0; i < 8; ++i) {
            int m = rsub + i;
            sT1[(wave * 4 + (col >> 5)) * FRAG_H + a_swz_idx(m, col & 31)] =
                (_Float16)sspf(acc[nt][i] + bi);
        }
    }
    __syncthreads();

    // GEMM2: [16,128] x fw2[128,128], +b, * cutoff
    v8f acc2[8] = {};
#pragma unroll
    for (int ks = 0; ks < 4; ++ks) {
        v16h a = load_frag(sT1 + (wave * 4 + ks) * FRAG_H);
#pragma unroll
        for (int nt = 0; nt < 8; ++nt) {
            v16h b = load_frag(fw2h + (ks * 8 + nt) * FRAG_H);
            acc2[nt] = wmma_f16(a, b, acc2[nt]);
        }
    }
#pragma unroll
    for (int nt = 0; nt < 8; ++nt) {
        int col = nt * 16 + col_lo;
        float bi = fb2[col];
#pragma unroll
        for (int i = 0; i < 8; ++i) {
            int r = wave * 16 + rsub + i;
            sW[r * 128 + col] = (acc2[nt][i] + bi) * sC[r];
        }
    }
    __syncthreads();

    // gather hf[ind_j]*W, scatter-add agg[ind_i]; 4 features/lane, 16 edges/wave
    int f4 = (tid & 31) * 4;
    int ebase = (tid >> 5) * 16;
    for (int e8 = 0; e8 < 16; ++e8) {
        int e = ebase + e8;
        int j = sJ[e], i = sI[e];
        float4 hv = *(const float4*)&hf[(size_t)j * NF + f4];
        float4 w = *(const float4*)&sW[e * 128 + f4];
        float* dst = &agg[(size_t)i * NF + f4];
        atomicAdd(dst + 0, hv.x * w.x);
        atomicAdd(dst + 1, hv.y * w.y);
        atomicAdd(dst + 2, hv.z * w.z);
        atomicAdd(dst + 3, hv.w * w.w);
    }
}

// ------- fused node update: h += ssp(agg@f2out+b)@lin + b -------
__global__ void __launch_bounds__(128)
node_update(const float* __restrict__ agg, const _Float16* __restrict__ f2outh,
            const float* __restrict__ f2out_b, const _Float16* __restrict__ linh,
            const float* __restrict__ lin_b, float* __restrict__ h, int nrows) {
    __shared__ __attribute__((aligned(16))) _Float16 sA[16 * FRAG_H];
    __shared__ __attribute__((aligned(16))) _Float16 sT[16 * FRAG_H];
    int tid = threadIdx.x;
    int row0 = blockIdx.x * 64;
    stage_a_tile(agg, row0, nrows, sA);
    __syncthreads();
    int wave = tid >> 5, lane = tid & 31;
    int col_lo = lane & 15;
    int rsub = (lane & 16) ? 8 : 0;

    v8f acc[8] = {};
#pragma unroll
    for (int ks = 0; ks < 4; ++ks) {
        v16h a = load_frag(sA + (wave * 4 + ks) * FRAG_H);
#pragma unroll
        for (int nt = 0; nt < 8; ++nt) {
            v16h b = load_frag(f2outh + (ks * 8 + nt) * FRAG_H);
            acc[nt] = wmma_f16(a, b, acc[nt]);
        }
    }
#pragma unroll
    for (int nt = 0; nt < 8; ++nt) {
        int col = nt * 16 + col_lo;
        float bi = f2out_b[col];
#pragma unroll
        for (int i = 0; i < 8; ++i) {
            int m = rsub + i;
            sT[(wave * 4 + (col >> 5)) * FRAG_H + a_swz_idx(m, col & 31)] =
                (_Float16)sspf(acc[nt][i] + bi);
        }
    }
    __syncthreads();

    v8f acc2[8] = {};
#pragma unroll
    for (int ks = 0; ks < 4; ++ks) {
        v16h a = load_frag(sT + (wave * 4 + ks) * FRAG_H);
#pragma unroll
        for (int nt = 0; nt < 8; ++nt) {
            v16h b = load_frag(linh + (ks * 8 + nt) * FRAG_H);
            acc2[nt] = wmma_f16(a, b, acc2[nt]);
        }
    }
#pragma unroll
    for (int nt = 0; nt < 8; ++nt) {
        int col = nt * 16 + col_lo;
        float bi = lin_b[col];
#pragma unroll
        for (int i = 0; i < 8; ++i) {
            int gr = row0 + wave * 16 + rsub + i;
            if (gr < nrows) h[gr * DIM + col] += acc2[nt][i] + bi;
        }
    }
}

// ------- output MLP + segment pool (tiny; VALU) -------
__global__ void __launch_bounds__(128)
out_pool(const float* __restrict__ h, const int* __restrict__ batch,
         const float* __restrict__ ow1, const float* __restrict__ ob1,
         const float* __restrict__ ow2, const float* __restrict__ ob2,
         const float* __restrict__ ow3, const float* __restrict__ ob3,
         float* __restrict__ out) {
    __shared__ float sH[128];
    __shared__ float sO1[64];
    __shared__ float sO2[32];
    int n = blockIdx.x;
    int t = threadIdx.x;
    sH[t] = h[(size_t)n * DIM + t];
    __syncthreads();
    if (t < 64) {
        float s = ob1[t];
        for (int k = 0; k < 128; ++k) s += sH[k] * ow1[k * 64 + t];
        sO1[t] = sspf(s);
    }
    __syncthreads();
    if (t < 32) {
        float s = ob2[t];
        for (int k = 0; k < 64; ++k) s += sO1[k] * ow2[k * 32 + t];
        sO2[t] = sspf(s);
    }
    __syncthreads();
    if (t == 0) {
        float s = ob3[0];
        for (int k = 0; k < 32; ++k) s += sO2[k] * ow3[k];
        atomicAdd(&out[batch[n]], s);
    }
}

extern "C" void kernel_launch(void* const* d_in, const int* in_sizes, int n_in,
                              void* d_out, int out_size, void* d_ws, size_t ws_size,
                              hipStream_t stream) {
    (void)in_sizes; (void)n_in; (void)out_size; (void)ws_size;
    const float* x       = (const float*)d_in[0];
    const float* dist    = (const float*)d_in[1];
    const int*   didx    = (const int*)d_in[2];
    const int*   batch   = (const int*)d_in[3];
    const float* win     = (const float*)d_in[4];
    const float* fw1     = (const float*)d_in[5];
    const float* fb1     = (const float*)d_in[6];
    const float* fw2     = (const float*)d_in[7];
    const float* fb2     = (const float*)d_in[8];
    const float* in2f_w  = (const float*)d_in[9];
    const float* f2out_w = (const float*)d_in[10];
    const float* f2out_b = (const float*)d_in[11];
    const float* lin_w   = (const float*)d_in[12];
    const float* lin_b   = (const float*)d_in[13];
    const float* ow1     = (const float*)d_in[14];
    const float* ob1     = (const float*)d_in[15];
    const float* ow2     = (const float*)d_in[16];
    const float* ob2     = (const float*)d_in[17];
    const float* ow3     = (const float*)d_in[18];
    const float* ob3     = (const float*)d_in[19];
    float* out = (float*)d_out;

    char* ws = (char*)d_ws;
    size_t off = 0;
    auto alloc = [&](size_t bytes) {
        char* p = ws + off;
        off += (bytes + 255) & ~(size_t)255;
        return p;
    };
    float*    h      = (float*)alloc((size_t)NN * DIM * 4);
    float*    hf     = (float*)alloc((size_t)NN * NF * 4);
    float*    agg    = (float*)alloc((size_t)NN * NF * 4);
    _Float16* winh   = (_Float16*)alloc((size_t)32 * FRAG_H * 2);
    _Float16* in2fh  = (_Float16*)alloc((size_t)NLAYERS * 32 * FRAG_H * 2);
    _Float16* fw1h   = (_Float16*)alloc((size_t)NLAYERS * 16 * FRAG_H * 2);
    _Float16* fw2h   = (_Float16*)alloc((size_t)NLAYERS * 32 * FRAG_H * 2);
    _Float16* f2outh = (_Float16*)alloc((size_t)NLAYERS * 32 * FRAG_H * 2);
    _Float16* linh   = (_Float16*)alloc((size_t)NLAYERS * 32 * FRAG_H * 2);

    // weight prep (f32 -> swizzled f16 fragments)
    int c1 = DIM * DIM;
    swz_weight_kernel<<<(c1 + 255) / 256, 256, 0, stream>>>(win, winh, 128, 1);
    int c6 = NLAYERS * DIM * DIM;
    swz_weight_kernel<<<(c6 + 255) / 256, 256, 0, stream>>>(in2f_w, in2fh, 128, NLAYERS);
    swz_weight_kernel<<<(c6 + 255) / 256, 256, 0, stream>>>(fw2, fw2h, 128, NLAYERS);
    swz_weight_kernel<<<(c6 + 255) / 256, 256, 0, stream>>>(f2out_w, f2outh, 128, NLAYERS);
    swz_weight_kernel<<<(c6 + 255) / 256, 256, 0, stream>>>(lin_w, linh, 128, NLAYERS);
    int cp = NLAYERS * 64 * 128;
    swz_fw1_kernel<<<(cp + 255) / 256, 256, 0, stream>>>(fw1, fw1h);

    int nodeBlocks = (NN + 63) / 64;  // 157
    int edgeBlocks = EE / 64;         // 5000

    node_gemm<<<nodeBlocks, 128, 0, stream>>>(x, winh, h, NN);

    for (int l = 0; l < NLAYERS; ++l) {
        node_gemm<<<nodeBlocks, 128, 0, stream>>>(h, in2fh + (size_t)l * 32 * FRAG_H, hf, NN);
        hipMemsetAsync(agg, 0, (size_t)NN * NF * sizeof(float), stream);
        edge_kernel<<<edgeBlocks, 128, 0, stream>>>(
            dist, didx,
            fw1h + (size_t)l * 16 * FRAG_H, fb1 + (size_t)l * NF,
            fw2h + (size_t)l * 32 * FRAG_H, fb2 + (size_t)l * NF,
            hf, agg);
        node_update<<<nodeBlocks, 128, 0, stream>>>(
            agg, f2outh + (size_t)l * 32 * FRAG_H, f2out_b + (size_t)l * DIM,
            linh + (size_t)l * 32 * FRAG_H, lin_b + (size_t)l * DIM, h, NN);
    }

    hipMemsetAsync(out, 0, GG * sizeof(float), stream);
    out_pool<<<NN, 128, 0, stream>>>(h, batch, ow1, ob1, ow2, ob2, ow3, ob3, out);
}
